// Seq2Seq_3917010174805
// MI455X (gfx1250) — matrix operand order repaired and checked by reading, not compiled, not run on hardware
//
#include <hip/hip_runtime.h>
#include <hip/hip_bf16.h>

// ---------------- problem constants (from reference setup) ----------------
#define HIDDEN  512
#define G4      2048          // 4*HIDDEN
#define T_SRC   512
#define T_TGT   256
#define BS      16            // batch rows per workgroup (= WMMA M)
#define NW      16            // waves per workgroup
#define NTHREADS 512
#define H_STRIDE_U32 265      // padded h-row stride in dwords (odd => bank-conflict-free A loads)
#define G_STRIDE 2052         // padded gates row stride in floats

typedef __attribute__((ext_vector_type(16))) __bf16 v16bf;
typedef __attribute__((ext_vector_type(8)))  float  v8f;

union BF16x16 { unsigned int u[8]; v16bf v; };

__device__ __forceinline__ unsigned short f2bf(float f) {
  unsigned int u = __float_as_uint(f);
  u += 0x7fffu + ((u >> 16) & 1u);          // round-to-nearest-even
  return (unsigned short)(u >> 16);
}
__device__ __forceinline__ float bf2f(unsigned short h) {
  return __uint_as_float(((unsigned int)h) << 16);
}
__device__ __forceinline__ float fsig(float x)  { return 1.0f / (1.0f + __expf(-x)); }
__device__ __forceinline__ float ftanh(float x) { return 2.0f * fsig(2.0f * x) - 1.0f; }

// one-time f32 -> bf16 conversion of both recurrent weight matrices into workspace
__global__ void cvt_bf16_kernel(const float* __restrict__ a, const float* __restrict__ b,
                                unsigned short* __restrict__ oa, unsigned short* __restrict__ ob,
                                int n) {
  int i = blockIdx.x * blockDim.x + threadIdx.x;
  if (i < n) { oa[i] = f2bf(a[i]); ob[i] = f2bf(b[i]); }
}

// gates[16,2048] = h[16,512] @ W^T  via v_wmma_f32_16x16x32_bf16
// Wt: bf16 row-major [2048][512] (row = gate unit j, col = k)
__device__ __forceinline__ void lstm_gemm(const unsigned short* __restrict__ Wt,
                                          const unsigned int* __restrict__ s_h32,
                                          float* __restrict__ s_gates,
                                          int wave, int lane) {
  const int n    = lane & 15;   // N column within tile / A row M
  const int half = lane >> 4;
  v8f acc[8] = {};
  const unsigned short* wp[8];
#pragma unroll
  for (int tt = 0; tt < 8; ++tt) {
    const int jt = wave * 8 + tt;                      // 16-wide N tile index (0..127)
    wp[tt] = Wt + (jt * 16 + n) * HIDDEN + half * 16;  // contiguous 16 halves per lane
  }
  for (int kb = 0; kb < HIDDEN; kb += 32) {
    // A fragment: contiguous k = 16*half + s (same order as B); h rows in LDS, odd-dword stride
    BF16x16 a;
#pragma unroll
    for (int p = 0; p < 8; ++p) {
      const int k0 = (p << 1) + (half << 4) + kb;  // even
      a.u[p] = s_h32[n * H_STRIDE_U32 + (k0 >> 1)];
    }
#pragma unroll
    for (int tt = 0; tt < 8; ++tt) {
      // B fragment: lanes 0-15 hold K=kb..kb+15, lanes 16-31 K=kb+16..kb+31 (32B contiguous)
      BF16x16 b;
      const uint4* p4 = reinterpret_cast<const uint4*>(wp[tt] + kb);
      uint4 lo = p4[0], hi = p4[1];
      b.u[0] = lo.x; b.u[1] = lo.y; b.u[2] = lo.z; b.u[3] = lo.w;
      b.u[4] = hi.x; b.u[5] = hi.y; b.u[6] = hi.z; b.u[7] = hi.w;
      acc[tt] = __builtin_amdgcn_wmma_f32_16x16x32_bf16(
          false, a.v, false, b.v, (short)0, acc[tt], false, false);
    }
  }
  // scatter C: VGPR r -> row (r + 8*half), col = jt*16 + n ; padded stride kills conflicts
#pragma unroll
  for (int tt = 0; tt < 8; ++tt) {
    const int col = (wave * 8 + tt) * 16 + n;
#pragma unroll
    for (int r = 0; r < 8; ++r)
      s_gates[(r + half * 8) * G_STRIDE + col] = acc[tt][r];
  }
}

// fused LSTM cell: gates += x*W_ih + bias ; update c (f32) and h (bf16)
__device__ __forceinline__ void lstm_elem(const float* __restrict__ s_gates,
                                          const float* __restrict__ s_wih,
                                          const float* __restrict__ s_bsum,
                                          const float* __restrict__ s_x,
                                          float* __restrict__ s_c,
                                          unsigned int* __restrict__ s_h32,
                                          int tid) {
  unsigned short* s_h16 = (unsigned short*)s_h32;
  const int j = tid;  // 0..511 == HIDDEN
  const float wi = s_wih[j],            wf = s_wih[HIDDEN + j];
  const float wg = s_wih[2*HIDDEN + j], wo = s_wih[3*HIDDEN + j];
  const float bi = s_bsum[j],            bf = s_bsum[HIDDEN + j];
  const float bg = s_bsum[2*HIDDEN + j], bo = s_bsum[3*HIDDEN + j];
#pragma unroll 4
  for (int m = 0; m < BS; ++m) {
    const float xv = s_x[m];
    const float* g = s_gates + m * G_STRIDE;
    const float ii = g[j]            + xv * wi + bi;
    const float ff = g[HIDDEN + j]   + xv * wf + bf;
    const float gg = g[2*HIDDEN + j] + xv * wg + bg;
    const float oo = g[3*HIDDEN + j] + xv * wo + bo;
    float c = fsig(ff) * s_c[m * HIDDEN + j] + fsig(ii) * ftanh(gg);
    float h = fsig(oo) * ftanh(c);
    s_c[m * HIDDEN + j] = c;
    s_h16[m * (H_STRIDE_U32 * 2) + j] = f2bf(h);
  }
}

__global__ __launch_bounds__(NTHREADS, 1)
void seq2seq_kernel(const float* __restrict__ input_seq,   // [256,512,1]
                    const float* __restrict__ We_ih,       // [2048,1]
                    const float* __restrict__ be_ih,
                    const float* __restrict__ be_hh,
                    const float* __restrict__ Wd_ih,       // [2048,1]
                    const float* __restrict__ bd_ih,
                    const float* __restrict__ bd_hh,
                    const float* __restrict__ W_fc,        // [1,512]
                    const float* __restrict__ b_fc,        // [1]
                    const unsigned short* __restrict__ Wenc,  // bf16 [2048,512]
                    const unsigned short* __restrict__ Wdec,  // bf16 [2048,512]
                    float* __restrict__ out)               // [256,256,1]
{
  __shared__ float        s_gates[BS * G_STRIDE];      // 131 KB
  __shared__ unsigned int s_h32[BS * H_STRIDE_U32];    // 17 KB (bf16 h, padded rows)
  __shared__ float        s_c[BS * HIDDEN];            // 32 KB
  __shared__ float        s_wih[G4];                   // 8 KB
  __shared__ float        s_bsum[G4];                  // 8 KB
  __shared__ float        s_wfc[HIDDEN];               // 2 KB
  __shared__ float        s_x[BS];                     // x_t (enc) / y feedback (dec)

  const int tid  = threadIdx.x;
  const int wave = tid >> 5;
  const int lane = tid & 31;
  const int bm0  = blockIdx.x * BS;
  const float bfc = b_fc[0];

  // ---- init LDS-resident state + encoder params ----
  for (int i = tid; i < BS * H_STRIDE_U32; i += NTHREADS) s_h32[i] = 0u;
  for (int i = tid; i < BS * HIDDEN;       i += NTHREADS) s_c[i]  = 0.0f;
  for (int j = tid; j < G4; j += NTHREADS) { s_wih[j] = We_ih[j]; s_bsum[j] = be_ih[j] + be_hh[j]; }
  s_wfc[tid & (HIDDEN - 1)] = W_fc[tid & (HIDDEN - 1)];
  __syncthreads();

  // ---- encoder scan ----
  for (int t = 0; t < T_SRC; ++t) {
    if (tid < BS) s_x[tid] = input_seq[(bm0 + tid) * T_SRC + t];
    lstm_gemm(Wenc, s_h32, s_gates, wave, lane);
    __syncthreads();
    lstm_elem(s_gates, s_wih, s_bsum, s_x, s_c, s_h32, tid);
    __syncthreads();
  }

  // ---- switch to decoder params; y0 = 0 ----
  for (int j = tid; j < G4; j += NTHREADS) { s_wih[j] = Wd_ih[j]; s_bsum[j] = bd_ih[j] + bd_hh[j]; }
  if (tid < BS) s_x[tid] = 0.0f;
  __syncthreads();

  // ---- autoregressive decoder ----
  for (int t = 0; t < T_TGT; ++t) {
    lstm_gemm(Wdec, s_h32, s_gates, wave, lane);
    __syncthreads();
    lstm_elem(s_gates, s_wih, s_bsum, s_x, s_c, s_h32, tid);
    __syncthreads();
    // pred[m] = h[m,:] . W_fc + b_fc ; one wave per batch row, shuffle-reduce
    if (wave < BS) {
      const int m = wave;
      const unsigned short* hrow = (const unsigned short*)s_h32 + m * (H_STRIDE_U32 * 2);
      float sum = 0.0f;
      for (int k = lane; k < HIDDEN; k += 32) sum += bf2f(hrow[k]) * s_wfc[k];
      for (int off = 16; off > 0; off >>= 1) sum += __shfl_xor(sum, off, 32);
      if (lane == 0) {
        const float pr = sum + bfc;
        s_x[m] = pr;                              // feedback (visible to EW after next barrier)
        out[(bm0 + m) * T_TGT + t] = pr;
      }
    }
  }
}

extern "C" void kernel_launch(void* const* d_in, const int* in_sizes, int n_in,
                              void* d_out, int out_size, void* d_ws, size_t ws_size,
                              hipStream_t stream) {
  (void)in_sizes; (void)n_in; (void)out_size; (void)ws_size;
  const float* input_seq = (const float*)d_in[0];
  const float* We_ih = (const float*)d_in[1];
  const float* We_hh = (const float*)d_in[2];
  const float* be_ih = (const float*)d_in[3];
  const float* be_hh = (const float*)d_in[4];
  const float* Wd_ih = (const float*)d_in[5];
  const float* Wd_hh = (const float*)d_in[6];
  const float* bd_ih = (const float*)d_in[7];
  const float* bd_hh = (const float*)d_in[8];
  const float* W_fc  = (const float*)d_in[9];
  const float* b_fc  = (const float*)d_in[10];
  float* out = (float*)d_out;

  // workspace: bf16 copies of the two 2048x512 recurrent matrices (2 MB each)
  const int NW_ELEM = G4 * HIDDEN;  // 1,048,576
  unsigned short* wenc = (unsigned short*)d_ws;
  unsigned short* wdec = wenc + NW_ELEM;

  cvt_bf16_kernel<<<NW_ELEM / 256, 256, 0, stream>>>(We_hh, Wd_hh, wenc, wdec, NW_ELEM);
  seq2seq_kernel<<<256 / BS, NTHREADS, 0, stream>>>(input_seq, We_ih, be_ih, be_hh,
                                                    Wd_ih, bd_ih, bd_hh, W_fc, b_fc,
                                                    wenc, wdec, out);
}
